// CrossAttention2D_67095979099223
// MI455X (gfx1250) — matrix operand order repaired and checked by reading, not compiled, not run on hardware
//
#include <hip/hip_runtime.h>
#include <stdint.h>

#define B_    8
#define C_    128
#define HW_   4096
#define NT    64      // key-tile size
#define MROWS 16      // query rows per wave
#define WAVES 8       // waves per workgroup (wave32)
#define LOG2E 1.44269504088896340736f

typedef __bf16 v16bf __attribute__((ext_vector_type(16)));
typedef float  v8f   __attribute__((ext_vector_type(8)));

union Frag {
  v16bf    bf;
  uint4    q[2];
  uint32_t u[8];
};

__device__ __forceinline__ uint16_t f2bf(float f) {
  uint32_t u = __float_as_uint(f);
  uint32_t r = (u + 0x7FFFu + ((u >> 16) & 1u)) >> 16;  // round-to-nearest-even
  return (uint16_t)r;
}

// gfx1250 async global->LDS copy, 16B per lane, tracked by ASYNCcnt.
__device__ __forceinline__ void async_b128(uint32_t lds_off, uint64_t gaddr) {
  asm volatile("global_load_async_to_lds_b128 %0, %1, off"
               :: "v"(lds_off), "v"(gaddr)
               : "memory");
}
__device__ __forceinline__ void wait_async0() {
  asm volatile("s_wait_asynccnt 0" ::: "memory");
}

// ---------------------------------------------------------------------------
// 1x1-conv projection: out[b,p,o] = sum_c x[b,c,p] * W[o,c] + bias[o]
// ---------------------------------------------------------------------------
__global__ __launch_bounds__(256)
void proj_kernel(const float* __restrict__ x,      // (B, C, HW)
                 const float* __restrict__ Wmat,   // (C_out, C_in)
                 const float* __restrict__ bias,   // (C)
                 uint16_t* __restrict__ out_bf,    // (B, HW, C)
                 float* __restrict__ out_f32,      // (B, HW, C) or nullptr
                 float scale)
{
  __shared__ float Wlds[C_][32];

  const int blk = blockIdx.x;                 // B * HW/64 blocks
  const int bb  = blk / (HW_ / 64);
  const int p0  = (blk % (HW_ / 64)) * 64;
  const int tid = threadIdx.x;
  const int p     = p0 + (tid & 63);
  const int oBase = (tid >> 6) * 32;

  float acc[32];
#pragma unroll
  for (int j = 0; j < 32; ++j) acc[j] = 0.f;

  for (int cb = 0; cb < 4; ++cb) {
    __syncthreads();
    for (int i = tid; i < C_ * 32; i += 256) {
      int o = i >> 5, cc = i & 31;
      Wlds[o][cc] = Wmat[o * C_ + cb * 32 + cc];
    }
    __syncthreads();
#pragma unroll 8
    for (int cc = 0; cc < 32; ++cc) {
      float xv = x[((size_t)bb * C_ + cb * 32 + cc) * HW_ + p];
#pragma unroll
      for (int j = 0; j < 32; ++j)
        acc[j] = fmaf(xv, Wlds[oBase + j][cc], acc[j]);
    }
  }

  size_t rowoff = ((size_t)bb * HW_ + p) * C_ + oBase;
#pragma unroll
  for (int j = 0; j < 32; ++j) {
    float v = acc[j] + bias[oBase + j];
    out_bf[rowoff + j] = f2bf(v * scale);
    if (out_f32) out_f32[rowoff + j] = v;
  }
}

// ---------------------------------------------------------------------------
// f32 -> bf16 bulk convert (for V), 4 elements/thread.
// ---------------------------------------------------------------------------
__global__ __launch_bounds__(256)
void cvt_bf16_kernel(const float* __restrict__ src, uint16_t* __restrict__ dst) {
  size_t i = ((size_t)blockIdx.x * 256 + threadIdx.x) * 4;
  float4 f = *(const float4*)(src + i);
  uint2 pk;
  pk.x = (uint32_t)f2bf(f.x) | ((uint32_t)f2bf(f.y) << 16);
  pk.y = (uint32_t)f2bf(f.z) | ((uint32_t)f2bf(f.w) << 16);
  *(uint2*)(dst + i) = pk;
}

// ---------------------------------------------------------------------------
// Flash attention with double-buffered async K/V staging.
// ---------------------------------------------------------------------------
__global__ __launch_bounds__(256)
void attn_kernel(const uint16_t* __restrict__ qbf,   // (B, HW, C) bf16, scale log2e/64
                 const uint16_t* __restrict__ kbf,   // (B, HW, C) bf16
                 const uint16_t* __restrict__ vbf,   // (B, C, HW) bf16
                 const float*    __restrict__ qres,  // (B, HW, C) f32 residual
                 float*          __restrict__ out)   // (B, C, HW) f32
{
  __shared__ uint16_t Klds[2][NT][C_];         // [buf][pos][c]  : 32 KB
  __shared__ uint16_t Vlds[2][C_][NT];         // [buf][c][pos]  : 32 KB
  __shared__ uint16_t Pscr[WAVES][MROWS][NT];  // per-wave P scratch: 16 KB

  const int tid  = threadIdx.x;
  const int wave = tid >> 5;
  const int lane = tid & 31;
  const int lrow = lane & 15;
  const int hi   = lane >> 4;

  const int blocksPerBatch = HW_ / (MROWS * WAVES);   // 32
  const int bb = blockIdx.x / blocksPerBatch;
  const int p0 = (blockIdx.x % blocksPerBatch) * (MROWS * WAVES) + wave * MROWS;

  // Per-thread staging slots (64B of K + 64B of V per thread per tile).
  const int kpos = tid >> 2, kseg = tid & 3;
  const int vc   = tid >> 1, vhalf = tid & 1;
  const uint16_t* kbase = kbf + ((size_t)bb * HW_ + kpos) * C_ + kseg * 32;
  const uint16_t* vbase = vbf + ((size_t)bb * C_ + vc) * HW_ + vhalf * 32;

  // Q A-fragments, resident for the whole kernel.
  Frag Aq[4];
  {
    const uint16_t* qrow = qbf + ((size_t)bb * HW_ + p0 + lrow) * C_;
#pragma unroll
    for (int cc = 0; cc < 4; ++cc) {
      Aq[cc].q[0] = *(const uint4*)(qrow + cc * 32 + hi * 8);
      Aq[cc].q[1] = *(const uint4*)(qrow + cc * 32 + 16 + hi * 8);
    }
  }

  float m_i[8], l_i[8];
#pragma unroll
  for (int r = 0; r < 8; ++r) { m_i[r] = -3.0e38f; l_i[r] = 0.f; }
  v8f O[8];
#pragma unroll
  for (int nc = 0; nc < 8; ++nc) O[nc] = v8f{0.f,0.f,0.f,0.f,0.f,0.f,0.f,0.f};

  // async-stage one K/V tile into buffer bi
#define STAGE(bi, k0)                                                          \
  do {                                                                         \
    const uint16_t* ks = kbase + (size_t)(k0) * C_;                            \
    uint16_t* kd = &Klds[bi][kpos][kseg * 32];                                 \
    _Pragma("unroll")                                                          \
    for (int i = 0; i < 4; ++i)                                                \
      async_b128((uint32_t)(uintptr_t)(kd + i * 8),                            \
                 (uint64_t)(uintptr_t)(ks + i * 8));                           \
    const uint16_t* vs = vbase + (k0);                                         \
    uint16_t* vd = &Vlds[bi][vc][vhalf * 32];                                  \
    _Pragma("unroll")                                                          \
    for (int i = 0; i < 4; ++i)                                                \
      async_b128((uint32_t)(uintptr_t)(vd + i * 8),                            \
                 (uint64_t)(uintptr_t)(vs + i * 8));                           \
  } while (0)

  STAGE(0, 0);
  int bi = 0;

  for (int k0 = 0; k0 < HW_; k0 += NT) {
    wait_async0();        // our tile writes done
    __syncthreads();      // everyone's tile writes done / prev buf free
    if (k0 + NT < HW_) STAGE(bi ^ 1, k0 + NT);

    // --- S = Q @ K^T (scores in log2 domain): 16 WMMAs
    v8f S[4];
#pragma unroll
    for (int ns = 0; ns < 4; ++ns) {
      v8f sacc = v8f{0.f,0.f,0.f,0.f,0.f,0.f,0.f,0.f};
#pragma unroll
      for (int cc = 0; cc < 4; ++cc) {
        Frag Bk;
        const uint16_t* kp = &Klds[bi][ns * 16 + lrow][cc * 32 + hi * 8];
        Bk.q[0] = *(const uint4*)(kp);
        Bk.q[1] = *(const uint4*)(kp + 16);
        sacc = __builtin_amdgcn_wmma_f32_16x16x32_bf16(
            false, Aq[cc].bf, false, Bk.bf, (short)0, sacc, false, false);
      }
      S[ns] = sacc;
    }

    // --- online softmax (exp2 domain)
    float alpha[8];
#pragma unroll
    for (int r = 0; r < 8; ++r) {
      float mx = fmaxf(fmaxf(S[0][r], S[1][r]), fmaxf(S[2][r], S[3][r]));
#pragma unroll
      for (int off = 1; off < 16; off <<= 1)
        mx = fmaxf(mx, __shfl_xor(mx, off, 32));
      float nm = fmaxf(m_i[r], mx);
      alpha[r] = exp2f(m_i[r] - nm);
      m_i[r] = nm;
    }
    float rs[8];
#pragma unroll
    for (int r = 0; r < 8; ++r) rs[r] = 0.f;
#pragma unroll
    for (int ns = 0; ns < 4; ++ns)
#pragma unroll
      for (int r = 0; r < 8; ++r) {
        float pv = exp2f(S[ns][r] - m_i[r]);
        S[ns][r] = pv;
        rs[r] += pv;
      }
#pragma unroll
    for (int r = 0; r < 8; ++r) {
#pragma unroll
      for (int off = 1; off < 16; off <<= 1)
        rs[r] += __shfl_xor(rs[r], off, 32);
      l_i[r] = l_i[r] * alpha[r] + rs[r];
    }
#pragma unroll
    for (int nc = 0; nc < 8; ++nc)
#pragma unroll
      for (int r = 0; r < 8; ++r) O[nc][r] *= alpha[r];

    // --- C-layout -> A-layout for P via wave-private LDS scratch
#pragma unroll
    for (int ns = 0; ns < 4; ++ns)
#pragma unroll
      for (int r = 0; r < 8; ++r)
        Pscr[wave][r + 8 * hi][ns * 16 + lrow] = f2bf(S[ns][r]);

    // --- O += P @ V : 16 WMMAs
#pragma unroll
    for (int kk = 0; kk < 2; ++kk) {
      Frag Ap;
      const uint16_t* pp = &Pscr[wave][lrow][kk * 32];
      Ap.q[0] = *(const uint4*)(pp + hi * 8);
      Ap.q[1] = *(const uint4*)(pp + 16 + hi * 8);
#pragma unroll
      for (int nc = 0; nc < 8; ++nc) {
        Frag Bv;
        const uint16_t* vp = &Vlds[bi][nc * 16 + lrow][kk * 32 + hi * 8];
        Bv.q[0] = *(const uint4*)(vp);
        Bv.q[1] = *(const uint4*)(vp + 16);
        O[nc] = __builtin_amdgcn_wmma_f32_16x16x32_bf16(
            false, Ap.bf, false, Bv.bf, (short)0, O[nc], false, false);
      }
    }
    bi ^= 1;
  }

  // --- epilogue: normalize, add residual q, store transposed to (B, C, HW)
#pragma unroll
  for (int nc = 0; nc < 8; ++nc) {
#pragma unroll
    for (int r = 0; r < 8; ++r) {
      int M = r + 8 * hi;
      int c = nc * 16 + lrow;
      size_t p = (size_t)(p0 + M);
      float val = O[nc][r] / l_i[r]
                + qres[((size_t)bb * HW_ + p) * C_ + c];
      out[(size_t)bb * C_ * HW_ + (size_t)c * HW_ + p] = val;
    }
  }
}

// ---------------------------------------------------------------------------
extern "C" void kernel_launch(void* const* d_in, const int* in_sizes, int n_in,
                              void* d_out, int out_size, void* d_ws, size_t ws_size,
                              hipStream_t stream) {
  const float* query = (const float*)d_in[0];
  const float* key   = (const float*)d_in[1];
  const float* value = (const float*)d_in[2];
  const float* Wq    = (const float*)d_in[3];
  const float* bq    = (const float*)d_in[4];
  const float* Wk    = (const float*)d_in[5];
  const float* bk    = (const float*)d_in[6];
  float* out = (float*)d_out;

  const size_t NELT = (size_t)B_ * HW_ * C_;           // 4.19M elements
  char* ws = (char*)d_ws;
  uint16_t* qbf  = (uint16_t*)ws;                      //  8 MB
  uint16_t* kbf  = (uint16_t*)(ws + NELT * 2);         //  8 MB
  float*    qres = (float*)   (ws + NELT * 4);         // 16 MB
  uint16_t* vbf  = (uint16_t*)(ws + NELT * 8);         //  8 MB

  dim3 pgrid(B_ * (HW_ / 64));
  proj_kernel<<<pgrid, 256, 0, stream>>>(query, Wq, bq, qbf, qres, LOG2E / 64.0f);
  proj_kernel<<<pgrid, 256, 0, stream>>>(key,   Wk, bk, kbf, nullptr, 1.0f);
  cvt_bf16_kernel<<<dim3((unsigned)(NELT / 1024)), 256, 0, stream>>>(value, vbf);

  dim3 agrid(B_ * (HW_ / (MROWS * WAVES)));
  attn_kernel<<<agrid, 256, 0, stream>>>(qbf, kbf, vbf, qres, out);
}
